// FFN_Shared_Plus_TaskLoRA_3023656976884
// MI455X (gfx1250) — matrix operand order repaired and checked by reading, not compiled, not run on hardware
//
#include <hip/hip_runtime.h>

// ---- types -----------------------------------------------------------------
typedef __bf16 bf16;
typedef __attribute__((ext_vector_type(16))) __bf16 v16b;
typedef __attribute__((ext_vector_type(8)))  __bf16 v8b;
typedef __attribute__((ext_vector_type(8)))  float  v8f;
typedef int v4i __attribute__((vector_size(16)));   // matches async-LDS builtin

// ---- problem dims ----------------------------------------------------------
#define TB     4
#define TN     4096
#define TD     768
#define TMID   3072
#define TE     8
#define TINNER 24
#define TECOLS (TE * TINNER)   // 192
#define TT     (TB * TN)       // 16384

// ---- CDNA5 async global->LDS path (guarded; falls back to reg prefetch) ----
#if defined(__has_builtin)
#if __has_builtin(__builtin_amdgcn_global_load_async_to_lds_b128) && \
    __has_builtin(__builtin_amdgcn_s_wait_asynccnt)
#define USE_ASYNC_LDS 1
#endif
#endif
#ifndef USE_ASYNC_LDS
#define USE_ASYNC_LDS 0
#endif

__device__ __forceinline__ float gelu_tanh(float x) {
    const float c0 = 0.7978845608028654f;   // sqrt(2/pi)
    const float c1 = 0.044715f;
    float t = tanhf(c0 * (x + c1 * x * x * x));
    return 0.5f * x * (1.0f + t);
}

// ---- prep kernels ----------------------------------------------------------
__global__ void k_fold_bf16(const float* __restrict__ a, const float* __restrict__ b,
                            bf16* __restrict__ o, int n) {
    int i = blockIdx.x * blockDim.x + threadIdx.x;
    if (i < n) o[i] = (bf16)(a[i] + b[i]);
}
__global__ void k_cvt_bf16(const float* __restrict__ a, bf16* __restrict__ o, int n) {
    int i = blockIdx.x * blockDim.x + threadIdx.x;
    if (i < n) o[i] = (bf16)a[i];
}
__global__ void k_add_f32(const float* __restrict__ a, const float* __restrict__ b,
                          float* __restrict__ o, int n) {
    int i = blockIdx.x * blockDim.x + threadIdx.x;
    if (i < n) o[i] = a[i] + b[i];
}
// W2m[d*192 + e*24 + i] = We2[e][d][i]   (We2 is [E, D, INNER])
__global__ void k_w2m(const float* __restrict__ we2, bf16* __restrict__ o) {
    int i = blockIdx.x * blockDim.x + threadIdx.x;
    if (i < TD * TECOLS) {
        int d = i / TECOLS;
        int j = i % TECOLS;
        int e = j / TINNER;
        int ii = j % TINNER;
        o[i] = (bf16)we2[((size_t)e * TD + d) * TINNER + ii];
    }
}

// ---- gate: noisy top-2 softmax -> dense [T,8] combine weights --------------
__global__ __launch_bounds__(256) void k_gate(const float* __restrict__ x,
                                              const float* __restrict__ noise,
                                              const float* __restrict__ Wg,
                                              const float* __restrict__ bg,
                                              float* __restrict__ wout) {
    int wave = threadIdx.x >> 5;
    int lane = threadIdx.x & 31;
    int t = blockIdx.x * 8 + wave;

    float acc[TE];
#pragma unroll
    for (int e = 0; e < TE; ++e) acc[e] = 0.f;

    const float* xt = x + (size_t)t * TD;
    for (int d = lane; d < TD; d += 32) {
        float xv = xt[d];
#pragma unroll
        for (int e = 0; e < TE; ++e) acc[e] += xv * Wg[e * TD + d];
    }
#pragma unroll
    for (int e = 0; e < TE; ++e) {
#pragma unroll
        for (int off = 16; off > 0; off >>= 1)
            acc[e] += __shfl_xor(acc[e], off, 32);
    }
    if (lane == 0) {
        float lg[TE];
#pragma unroll
        for (int e = 0; e < TE; ++e)
            lg[e] = acc[e] + bg[e] + 0.1f * noise[(size_t)t * TE + e];
        int i1 = 0; float v1 = lg[0];
#pragma unroll
        for (int e = 1; e < TE; ++e) if (lg[e] > v1) { v1 = lg[e]; i1 = e; }
        int i2 = -1; float v2 = -3.4e38f;
#pragma unroll
        for (int e = 0; e < TE; ++e) if (e != i1 && lg[e] > v2) { v2 = lg[e]; i2 = e; }
        float p2 = __expf(v2 - v1);
        float s = 1.f + p2;
        float g1 = 1.f / s, g2 = p2 / s;
#pragma unroll
        for (int e = 0; e < TE; ++e)
            wout[(size_t)t * TE + e] = (e == i1) ? g1 : ((e == i2) ? g2 : 0.f);
    }
}

// ---- tiled WMMA GEMM: C[M,N] = A[M,K] * B[N,K]^T  (+ epilogue) -------------
// Double-buffered LDS, software-pipelined: next tile's global traffic is issued
// before the current tile's WMMAs (async-to-LDS when available).
enum { EPI_GELU_BF16 = 0, EPI_BIAS_F32 = 1, EPI_GATE_GELU_BF16 = 2, EPI_MOE_ADD_F32 = 3 };

template <int EPI>
__global__ __launch_bounds__(256) void k_gemm(const bf16* __restrict__ A,
                                              const bf16* __restrict__ Bw,
                                              const float* __restrict__ bias,
                                              bf16* __restrict__ outB,
                                              float* __restrict__ outF,
                                              const float* __restrict__ wgate,
                                              int M, int N, int K) {
    constexpr int BM = 128, BN = 128, BK = 32, LDT = BK + 8;   // pad -> conflict-free
    __shared__ __align__(16) bf16 sAbuf[2 * BM * LDT];
    __shared__ __align__(16) bf16 sBbuf[2 * BN * LDT];

    const int tid  = threadIdx.x;
    const int lane = tid & 31;
    const int wave = tid >> 5;
    const int wm   = wave >> 1;       // 0..3 : 32-row band
    const int wn   = wave & 1;        // 0..1 : 64-col band

    const int bm = blockIdx.y * BM;
    const int bn = blockIdx.x * BN;

    const int lrow = tid >> 1;              // 0..127
    const int lcol = (tid & 1) * 16;        // 0 or 16
    const bool gnv = (bn + lrow) < N;       // B row valid (ragged N)

    v8f acc[2][4];
#pragma unroll
    for (int i = 0; i < 2; ++i)
#pragma unroll
        for (int j = 0; j < 4; ++j) acc[i][j] = (v8f){};

    // fragment math from LDS buffer `cur`
    auto compute = [&](int cur) {
        const bf16* cA = &sAbuf[cur * BM * LDT];
        const bf16* cB = &sBbuf[cur * BN * LDT];
        // A frag: lane row = m, 16 consecutive K at kh*16
        v16b af[2];
#pragma unroll
        for (int i = 0; i < 2; ++i) {
            int row = wm * 32 + i * 16 + (lane & 15);
            int kh  = lane >> 4;
            const bf16* p = &cA[row * LDT + kh * 16];
            union { v16b v; v8b h[2]; } u;
            u.h[0] = *(const v8b*)p;
            u.h[1] = *(const v8b*)(p + 8);
            af[i] = u.v;
        }
        // B frag: lane col = n, 16 consecutive K at (lane/16)*16
        v16b bfr[4];
#pragma unroll
        for (int j = 0; j < 4; ++j) {
            int row = wn * 64 + j * 16 + (lane & 15);
            int kb  = (lane >> 4) * 16;
            const bf16* p = &cB[row * LDT + kb];
            union { v16b v; v8b h[2]; } u;
            u.h[0] = *(const v8b*)p;
            u.h[1] = *(const v8b*)(p + 8);
            bfr[j] = u.v;
        }
#pragma unroll
        for (int i = 0; i < 2; ++i)
#pragma unroll
            for (int j = 0; j < 4; ++j)
                acc[i][j] = __builtin_amdgcn_wmma_f32_16x16x32_bf16(
                    false, af[i], false, bfr[j], (short)0, acc[i][j], false, false);
    };

#if USE_ASYNC_LDS
    // ---- async global->LDS pipeline (ASYNCcnt-tracked, no VGPR staging) ----
    auto async16 = [&](bf16* l, const bf16* g) {
        __builtin_amdgcn_global_load_async_to_lds_b128(
            (__attribute__((address_space(1))) v4i*)g,
            (__attribute__((address_space(3))) v4i*)l, 0, 0);
    };
    auto issue_tile = [&](int buf, int kk) {
        bf16* dA = &sAbuf[buf * BM * LDT + lrow * LDT + lcol];
        const bf16* ga = A + (size_t)(bm + lrow) * K + kk + lcol;
        async16(dA, ga);
        async16(dA + 8, ga + 8);
        if (gnv) {
            bf16* dB = &sBbuf[buf * BN * LDT + lrow * LDT + lcol];
            const bf16* gb = Bw + (size_t)(bn + lrow) * K + kk + lcol;
            async16(dB, gb);
            async16(dB + 8, gb + 8);
        }
    };
    // ragged-N rows of B stay zero in both buffers (never rewritten)
    if (!gnv) {
        v8b z = {};
#pragma unroll
        for (int buf = 0; buf < 2; ++buf) {
            *(v8b*)&sBbuf[buf * BN * LDT + lrow * LDT + lcol]     = z;
            *(v8b*)&sBbuf[buf * BN * LDT + lrow * LDT + lcol + 8] = z;
        }
    }
    issue_tile(0, 0);
    int cur = 0;
    for (int kk = 0; kk < K; kk += BK) {
        __builtin_amdgcn_s_wait_asynccnt(0);   // this wave's tile `cur` landed
        __syncthreads();                       // everyone's part of `cur` landed
        if (kk + BK < K) issue_tile(cur ^ 1, kk + BK);  // overlap with compute
        compute(cur);
        cur ^= 1;
    }
#else
    // ---- register-prefetch double buffer (plain loads, still overlapped) ---
    v8b ra0, ra1, rb0, rb1;
    auto gload = [&](int kk) {
        const bf16* ga = A + (size_t)(bm + lrow) * K + kk + lcol;
        ra0 = *(const v8b*)ga;
        ra1 = *(const v8b*)(ga + 8);
        if (gnv) {
            const bf16* gb = Bw + (size_t)(bn + lrow) * K + kk + lcol;
            rb0 = *(const v8b*)gb;
            rb1 = *(const v8b*)(gb + 8);
        } else {
            rb0 = (v8b){};
            rb1 = (v8b){};
        }
    };
    gload(0);
    int cur = 0;
    for (int kk = 0; kk < K; kk += BK) {
        bf16* dA = &sAbuf[cur * BM * LDT + lrow * LDT + lcol];
        bf16* dB = &sBbuf[cur * BN * LDT + lrow * LDT + lcol];
        *(v8b*)dA       = ra0;
        *(v8b*)(dA + 8) = ra1;
        *(v8b*)dB       = rb0;
        *(v8b*)(dB + 8) = rb1;
        __syncthreads();
        if (kk + BK < K) gload(kk + BK);       // in flight during WMMAs
        compute(cur);
        cur ^= 1;
    }
#endif

    // epilogue: C layout -> VGPR r maps to M = r + 8*(lane>=16); N = lane%16
#pragma unroll
    for (int i = 0; i < 2; ++i) {
        int mbase = bm + wm * 32 + i * 16 + 8 * (lane >> 4);
#pragma unroll
        for (int j = 0; j < 4; ++j) {
            int n_g = bn + wn * 64 + j * 16 + (lane & 15);
            if (n_g < N) {
#pragma unroll
                for (int r = 0; r < 8; ++r) {
                    int m_g = mbase + r;
                    float v = acc[i][j][r];
                    if (EPI == EPI_GELU_BF16) {
                        v = gelu_tanh(v + bias[n_g]);
                        outB[(size_t)m_g * N + n_g] = (bf16)v;
                    } else if (EPI == EPI_BIAS_F32) {
                        outF[(size_t)m_g * N + n_g] = v + bias[n_g];
                    } else if (EPI == EPI_GATE_GELU_BF16) {
                        v = gelu_tanh(v + bias[n_g]) * wgate[(size_t)m_g * TE + (n_g / TINNER)];
                        outB[(size_t)m_g * N + n_g] = (bf16)v;
                    } else {  // EPI_MOE_ADD_F32 : bias = be2 [E, D]
                        float bb = 0.f;
#pragma unroll
                        for (int e = 0; e < TE; ++e)
                            bb += wgate[(size_t)m_g * TE + e] * bias[e * TD + n_g];
                        outF[(size_t)m_g * N + n_g] += v + bb;
                    }
                }
            }
        }
    }
}

// ---- launch ----------------------------------------------------------------
extern "C" void kernel_launch(void* const* d_in, const int* in_sizes, int n_in,
                              void* d_out, int out_size, void* d_ws, size_t ws_size,
                              hipStream_t stream) {
    (void)in_sizes; (void)n_in; (void)out_size; (void)ws_size;
    const float* x    = (const float*)d_in[0];
    const float* nois = (const float*)d_in[1];
    const float* W1   = (const float*)d_in[2];
    const float* b1   = (const float*)d_in[3];
    const float* W2   = (const float*)d_in[4];
    const float* b2   = (const float*)d_in[5];
    const float* dW1  = (const float*)d_in[6];
    const float* db1  = (const float*)d_in[7];
    const float* dW2  = (const float*)d_in[8];
    const float* db2  = (const float*)d_in[9];
    const float* Wg   = (const float*)d_in[10];
    const float* bg   = (const float*)d_in[11];
    const float* We1  = (const float*)d_in[12];
    const float* be1  = (const float*)d_in[13];
    const float* We2  = (const float*)d_in[14];
    const float* be2  = (const float*)d_in[15];
    float* out = (float*)d_out;

    // workspace carve-up (256B aligned)
    char*  ws  = (char*)d_ws;
    size_t off = 0;
    auto carve = [&](size_t bytes) -> void* {
        void* p = ws + off;
        off = (off + bytes + 255) & ~(size_t)255;
        return p;
    };
    bf16*  xb   = (bf16*)carve((size_t)TT * TD * 2);
    bf16*  H    = (bf16*)carve((size_t)TT * TMID * 2);
    bf16*  G    = (bf16*)carve((size_t)TT * TECOLS * 2);
    bf16*  W1e  = (bf16*)carve((size_t)TMID * TD * 2);
    bf16*  W2e  = (bf16*)carve((size_t)TD * TMID * 2);
    bf16*  We1c = (bf16*)carve((size_t)TECOLS * TD * 2);
    bf16*  W2m  = (bf16*)carve((size_t)TD * TECOLS * 2);
    float* wg   = (float*)carve((size_t)TT * TE * 4);
    float* b1e  = (float*)carve((size_t)TMID * 4);
    float* b2e  = (float*)carve((size_t)TD * 4);

    const int THR = 256;
    auto blks = [](size_t n) { return (int)((n + 255) / 256); };

    k_cvt_bf16 <<<blks((size_t)TT * TD),   THR, 0, stream>>>(x,   xb,   TT * TD);
    k_fold_bf16<<<blks((size_t)TMID * TD), THR, 0, stream>>>(W1, dW1, W1e, TMID * TD);
    k_fold_bf16<<<blks((size_t)TD * TMID), THR, 0, stream>>>(W2, dW2, W2e, TD * TMID);
    k_cvt_bf16 <<<blks((size_t)TECOLS*TD), THR, 0, stream>>>(We1, We1c, TECOLS * TD);
    k_w2m      <<<blks((size_t)TD*TECOLS), THR, 0, stream>>>(We2, W2m);
    k_add_f32  <<<blks(TMID),              THR, 0, stream>>>(b1, db1, b1e, TMID);
    k_add_f32  <<<blks(TD),                THR, 0, stream>>>(b2, db2, b2e, TD);
    k_gate     <<<TT / 8,                  THR, 0, stream>>>(x, nois, Wg, bg, wg);

    // H = gelu(X @ W1e^T + b1e)              [T, MID], K = D
    dim3 g1(TMID / 128, TT / 128);
    k_gemm<EPI_GELU_BF16><<<g1, THR, 0, stream>>>(xb, W1e, b1e, H, nullptr, nullptr,
                                                  TT, TMID, TD);
    // out = H @ W2e^T + b2e                  [T, D],  K = MID
    dim3 g2(TD / 128, TT / 128);
    k_gemm<EPI_BIAS_F32><<<g2, THR, 0, stream>>>(H, W2e, b2e, nullptr, out, nullptr,
                                                 TT, TD, TMID);
    // G = gelu(X @ We1c^T + be1) * gate      [T, 192], K = D
    dim3 g3((TECOLS + 127) / 128, TT / 128);
    k_gemm<EPI_GATE_GELU_BF16><<<g3, THR, 0, stream>>>(xb, We1c, be1, G, nullptr, wg,
                                                       TT, TECOLS, TD);
    // out += G @ W2m^T + sum_e w*be2         [T, D],  K = 192
    dim3 g4(TD / 128, TT / 128);
    k_gemm<EPI_MOE_ADD_F32><<<g4, THR, 0, stream>>>(G, W2m, be2, nullptr, out, wg,
                                                    TT, TD, TECOLS);
}
